// AttentionGate_Wavelet_14697378087594
// MI455X (gfx1250) — matrix operand order repaired and checked by reading, not compiled, not run on hardware
//
#include <hip/hip_runtime.h>

typedef __attribute__((ext_vector_type(16))) __bf16 v16bf;
typedef __attribute__((ext_vector_type(8)))  float  v8f;

#define DEVFN static __device__ __forceinline__

union BfPack {
  unsigned short s[16];
  uint4 q[2];
  v16bf v;
};

DEVFN unsigned short f2bf(float f){
  unsigned int u = __float_as_uint(f);
  unsigned int r = u + 0x7FFFu + ((u >> 16) & 1u);
  return (unsigned short)(r >> 16);
}

DEVFN v8f wmma_bf16(v16bf a, v16bf b, v8f c){
  return __builtin_amdgcn_wmma_f32_16x16x32_bf16(false, a, false, b, (short)0, c, false, false);
}

// ---- WMMA operand loaders (contiguous k order k = 16*(lane/16) + s for A and B, wave32) ----

// A operand 16x32 (MxK) from row-major [M][K] bf16 storage.
// lane<16: M=lane, K = k0..k0+15; lane>=16: M=lane-16, K = k0+16..k0+31
DEVFN v16bf load_A_rowmajor(const unsigned short* base, int ld, int m0, int k0, int lane){
  int hl  = lane >> 4;
  int row = m0 + (lane & 15);
  const unsigned short* p = base + (size_t)row * ld + k0 + hl * 16;
  BfPack t;
  t.q[0] = *reinterpret_cast<const uint4*>(p);
  t.q[1] = *reinterpret_cast<const uint4*>(p + 8);
  return t.v;
}

// A operand from K-major storage [K][M] bf16 (i.e. the transpose lives in memory).
DEVFN v16bf load_A_kmajor(const unsigned short* base, int ldm, int m0, int k0, int lane){
  int hl  = lane >> 4;
  int col = m0 + (lane & 15);
  BfPack t;
  const unsigned short* p = base + (size_t)(k0 + hl * 16) * ldm + col;
#pragma unroll
  for (int i = 0; i < 8; ++i) t.s[i] = p[(size_t)i * ldm];
  const unsigned short* p2 = base + (size_t)(k0 + 8 + hl * 16) * ldm + col;
#pragma unroll
  for (int i = 0; i < 8; ++i) t.s[8 + i] = p2[(size_t)i * ldm];
  return t.v;
}

// B operand 32x16 (KxN) from row-major [K][N] bf16.
// lane<16: N=lane, K=k0..k0+15 ; lane>=16: N=lane-16, K=k0+16..k0+31
DEVFN v16bf load_B_kmajor(const unsigned short* base, int ldn, int n0, int k0, int lane){
  int hl  = lane >> 4;
  int col = n0 + (lane & 15);
  BfPack t;
  const unsigned short* p = base + (size_t)(k0 + hl * 16) * ldn + col;
#pragma unroll
  for (int i = 0; i < 16; ++i) t.s[i] = p[(size_t)i * ldn];
  return t.v;
}

// B operand where element(k,col) = exp(A[k][col] - cmax[col])  (column softmax numerator)
DEVFN v16bf load_B_expcol(const float* A, const float* cmax, int n0, int k0, int lane){
  int hl  = lane >> 4;
  int col = n0 + (lane & 15);
  float cm = cmax[col];
  BfPack t;
  const float* p = A + (size_t)(k0 + hl * 16) * 1024 + col;
#pragma unroll
  for (int i = 0; i < 16; ++i) t.s[i] = f2bf(__expf(p[(size_t)i * 1024] - cm));
  return t.v;
}

// B operand where element(k,col) = exp(A[col][k] - rmax[col])  (row softmax numerator, transposed read)
DEVFN v16bf load_B_exprow(const float* A, const float* rmax, int n0, int k0, int lane){
  int hl  = lane >> 4;
  int col = n0 + (lane & 15);
  float rm = rmax[col];
  BfPack t;
  const float* p = A + (size_t)col * 1024 + (k0 + hl * 16);
#pragma unroll
  for (int i = 0; i < 16; ++i) t.s[i] = f2bf(__expf(p[i] - rm));
  return t.v;
}

DEVFN float sigmoidf(float x){ return 1.0f / (1.0f + __expf(-x)); }
DEVFN float lreluf(float x){ return x >= 0.0f ? x : 0.2f * x; }

// =====================================================================
// Kernel 0: pack weights to bf16 (W_lin row-major [d][c]; conv: [tap][o][ci])
// =====================================================================
__global__ void k_convw(const float* __restrict__ Wlin,
                        const float* __restrict__ w1, const float* __restrict__ w2,
                        unsigned short* __restrict__ Wlb,
                        unsigned short* __restrict__ w1b, unsigned short* __restrict__ w2b){
  int idx = blockIdx.x * blockDim.x + threadIdx.x;
  const int n1 = 128 * 128;
  const int n2 = 9 * 128 * 384;
  if (idx < n1){
    Wlb[idx] = f2bf(Wlin[idx]);
    return;
  }
  idx -= n1;
  if (idx < 2 * n2){
    const float* src = (idx < n2) ? w1 : w2;
    unsigned short* dst = (idx < n2) ? w1b : w2b;
    int t = (idx < n2) ? idx : idx - n2;
    int tap = t / (128 * 384);
    int rem = t - tap * (128 * 384);
    int o  = rem / 384;
    int ci = rem - o * 384;
    int kh = tap / 3, kw = tap - kh * 3;
    dst[t] = f2bf(src[((o * 384 + ci) * 3 + kh) * 3 + kw]);
  }
}

// =====================================================================
// Kernel 1: bilinear 64->32 (exact 2x2 average for this scale) + bf16 copy
// =====================================================================
__global__ void k_down(const float* __restrict__ fr1, const float* __restrict__ fr2,
                       float* __restrict__ f1s, float* __restrict__ f2s,
                       unsigned short* __restrict__ f1b, unsigned short* __restrict__ f2b){
  int idx = blockIdx.x * blockDim.x + threadIdx.x; // 2*32*128*1024 = 2^23
  int p = idx & 1023;
  int j = p & 31, i = p >> 5;
  int c = (idx >> 10) & 127;
  int b = (idx >> 17) & 31;
  int f = idx >> 22;
  const float* src = (f ? fr2 : fr1) + ((size_t)b * 128 + c) * 4096;
  float v = 0.25f * (src[(2 * i) * 64 + 2 * j] + src[(2 * i + 1) * 64 + 2 * j] +
                     src[(2 * i) * 64 + 2 * j + 1] + src[(2 * i + 1) * 64 + 2 * j + 1]);
  size_t o = ((size_t)b * 128 + c) * 1024 + p;
  (f ? f2s : f1s)[o] = v;
  (f ? f2b : f1b)[o] = f2bf(v);
}

// =====================================================================
// Kernel 2: g[b][d][n] = sum_c W_lin[d][c] * f1[b][c][n]   (M=128,N=1024,K=128)
// =====================================================================
__global__ void k_lin(const unsigned short* __restrict__ Wb,
                      const unsigned short* __restrict__ f1b,
                      unsigned short* __restrict__ g){
  int lane  = threadIdx.x & 31;
  int wtile = blockIdx.x * (blockDim.x >> 5) + (threadIdx.x >> 5); // 32*8*64
  int nt = wtile & 63;
  int mt = (wtile >> 6) & 7;
  int b  = wtile >> 9;
  const unsigned short* Bp = f1b + (size_t)b * 128 * 1024;
  v8f acc = {0.f,0.f,0.f,0.f,0.f,0.f,0.f,0.f};
#pragma unroll
  for (int k = 0; k < 128; k += 32){
    v16bf a  = load_A_rowmajor(Wb, 128, mt * 16, k, lane);
    v16bf bb = load_B_kmajor(Bp, 1024, nt * 16, k, lane);
    acc = wmma_bf16(a, bb, acc);
  }
  unsigned short* D = g + (size_t)b * 128 * 1024;
  int col   = nt * 16 + (lane & 15);
  int rbase = mt * 16 + ((lane >> 4) << 3);
#pragma unroll
  for (int r = 0; r < 8; ++r) D[(size_t)(rbase + r) * 1024 + col] = f2bf(acc[r]);
}

// =====================================================================
// Kernel 3: A[b][n][m] = sum_c g[b][c][n] * f2[b][c][m]  (M=N=1024,K=128), f32 out
// =====================================================================
__global__ void k_attA(const unsigned short* __restrict__ g,
                       const unsigned short* __restrict__ f2b,
                       float* __restrict__ Amat){
  int lane  = threadIdx.x & 31;
  int wtile = blockIdx.x * (blockDim.x >> 5) + (threadIdx.x >> 5); // 32*64*64
  int nt = wtile & 63;          // m tile
  int mt = (wtile >> 6) & 63;   // n tile
  int b  = wtile >> 12;
  const unsigned short* Ap = g   + (size_t)b * 128 * 1024;
  const unsigned short* Bp = f2b + (size_t)b * 128 * 1024;
  v8f acc = {0.f,0.f,0.f,0.f,0.f,0.f,0.f,0.f};
#pragma unroll
  for (int k = 0; k < 128; k += 32){
    v16bf a  = load_A_kmajor(Ap, 1024, mt * 16, k, lane);
    v16bf bb = load_B_kmajor(Bp, 1024, nt * 16, k, lane);
    acc = wmma_bf16(a, bb, acc);
  }
  float* D = Amat + ((size_t)b << 20);
  int col   = nt * 16 + (lane & 15);
  int rbase = mt * 16 + ((lane >> 4) << 3);
#pragma unroll
  for (int r = 0; r < 8; ++r) D[(size_t)(rbase + r) * 1024 + col] = acc[r];
}

// =====================================================================
// Kernel 4a/4b: online softmax statistics (max + sum of exp), per column / per row
// =====================================================================
__global__ void k_red_col(const float* __restrict__ A, float* __restrict__ cmax, float* __restrict__ csum){
  int idx = blockIdx.x * blockDim.x + threadIdx.x; // b*1024 + m
  int b = idx >> 10, m = idx & 1023;
  const float* p = A + ((size_t)b << 20) + m;
  float mx = -1e30f, s = 0.f;
  for (int n = 0; n < 1024; ++n){
    float v = p[(size_t)n << 10];
    if (v > mx){ s = s * __expf(mx - v) + 1.f; mx = v; }
    else        { s += __expf(v - mx); }
  }
  cmax[idx] = mx; csum[idx] = s;
}

__global__ void k_red_row(const float* __restrict__ A, float* __restrict__ rmax, float* __restrict__ rsum){
  int idx = blockIdx.x * blockDim.x + threadIdx.x; // b*1024 + n
  int b = idx >> 10, n = idx & 1023;
  const float* p = A + ((size_t)b << 20) + ((size_t)n << 10);
  float mx = -1e30f, s = 0.f;
  for (int m = 0; m < 1024; ++m){
    float v = p[m];
    if (v > mx){ s = s * __expf(mx - v) + 1.f; mx = v; }
    else        { s += __expf(v - mx); }
  }
  rmax[idx] = mx; rsum[idx] = s;
}

// =====================================================================
// Kernel 5: attention apply, exp() fused into the B operand loader.
// MODE 0: f2_att[b][c][m] = (sum_n f1[c][n] exp(A[n][m]-cmax[m])) / csum[m]
// MODE 1: f1_att[b][c][j] = (sum_i f2[c][i] exp(A[j][i]-rmax[j])) / rsum[j]
// =====================================================================
template <int MODE>
__global__ void k_att_apply(const unsigned short* __restrict__ Fb,
                            const float* __restrict__ Amat,
                            const float* __restrict__ vmax,
                            const float* __restrict__ vsum,
                            float* __restrict__ out){
  int lane  = threadIdx.x & 31;
  int wtile = blockIdx.x * (blockDim.x >> 5) + (threadIdx.x >> 5); // 32*8*64
  int nt = wtile & 63;
  int mt = (wtile >> 6) & 7;
  int b  = wtile >> 9;
  const unsigned short* Ap = Fb + (size_t)b * 128 * 1024;
  const float* AM = Amat + ((size_t)b << 20);
  const float* mx = vmax + (size_t)b * 1024;
  const float* sm = vsum + (size_t)b * 1024;
  v8f acc = {0.f,0.f,0.f,0.f,0.f,0.f,0.f,0.f};
  for (int k = 0; k < 1024; k += 32){
    if (k + 32 < 1024)
      __builtin_prefetch((const void*)(AM + (size_t)(k + 32) * 1024 + nt * 16), 0, 1);
    v16bf a  = load_A_rowmajor(Ap, 1024, mt * 16, k, lane);
    v16bf bb = MODE ? load_B_exprow(AM, mx, nt * 16, k, lane)
                    : load_B_expcol(AM, mx, nt * 16, k, lane);
    acc = wmma_bf16(a, bb, acc);
  }
  float* D  = out + (size_t)b * 128 * 1024;
  int col   = nt * 16 + (lane & 15);
  float inv = 1.0f / sm[col];
  int rbase = mt * 16 + ((lane >> 4) << 3);
#pragma unroll
  for (int r = 0; r < 8; ++r) D[(size_t)(rbase + r) * 1024 + col] = acc[r] * inv;
}

// =====================================================================
// Kernel 6: spatial gates m1,m2 per (b,pixel)
// =====================================================================
__global__ void k_m1m2(const float* __restrict__ f1a, const float* __restrict__ f2a,
                       const float* __restrict__ g1,  const float* __restrict__ g2,
                       float* __restrict__ m1, float* __restrict__ m2){
  int idx = blockIdx.x * blockDim.x + threadIdx.x; // 32*1024
  int b = idx >> 10, p = idx & 1023;
  const float* p1 = f1a + (size_t)b * 128 * 1024 + p;
  const float* p2 = f2a + (size_t)b * 128 * 1024 + p;
  float d1 = 0.f, d2 = 0.f;
  for (int c = 0; c < 128; ++c){
    float x1 = p1[(size_t)c << 10], x2 = p2[(size_t)c << 10];
    d1 += x1 * g1[c] + x2 * g1[128 + c];
    d2 += x1 * g2[c] + x2 * g2[128 + c];
  }
  m1[idx] = sigmoidf(d1);
  m2[idx] = sigmoidf(d2);
}

// Kernel 7: pooled[b][ch<256] = mean over 1024 pixels of cat
__global__ void k_pool(const float* __restrict__ f1a, const float* __restrict__ f2a,
                       float* __restrict__ pooled){
  int idx = blockIdx.x * blockDim.x + threadIdx.x; // 32*256
  int b = idx >> 8, ch = idx & 255;
  const float* src = (ch < 128 ? f1a + ((size_t)b * 128 + ch) * 1024
                               : f2a + ((size_t)b * 128 + (ch - 128)) * 1024);
  float s = 0.f;
  for (int p = 0; p < 1024; ++p) s += src[p];
  pooled[idx] = s * (1.0f / 1024.0f);
}

// Kernel 8: SE gate: out[b][c] = sigmoid(W2 * lrelu(W1*pooled + b1) + b2). grid = B.
__global__ void k_se(const float* __restrict__ pooled, int pdim,
                     const float* __restrict__ w1, const float* __restrict__ b1,
                     const float* __restrict__ w2, const float* __restrict__ b2,
                     int odim, float* __restrict__ out){
  __shared__ float h[16];
  int b = blockIdx.x;
  const float* pl = pooled + (size_t)b * pdim;
  if (threadIdx.x < 16){
    float s = b1[threadIdx.x];
    const float* w = w1 + (size_t)threadIdx.x * pdim;
    for (int j = 0; j < pdim; ++j) s += w[j] * pl[j];
    h[threadIdx.x] = lreluf(s);
  }
  __syncthreads();
  for (int c = threadIdx.x; c < odim; c += blockDim.x){
    float s = b2[c];
    const float* w = w2 + (size_t)c * 16;
#pragma unroll
    for (int k = 0; k < 16; ++k) s += w[k] * h[k];
    out[(size_t)b * odim + c] = sigmoidf(s);
  }
}

// Kernel 9: Haar DWT of f1_att / f2_att -> [b][4*128][16][16]
__global__ void k_dwt(const float* __restrict__ f1a, const float* __restrict__ f2a,
                      float* __restrict__ f1f, float* __restrict__ f2f){
  int idx = blockIdx.x * blockDim.x + threadIdx.x; // 2*32*128*256 = 2^21
  int j = idx & 15, i = (idx >> 4) & 15;
  int c = (idx >> 8) & 127;
  int b = (idx >> 15) & 31;
  int f = idx >> 20;
  const float* src = (f ? f2a : f1a) + ((size_t)b * 128 + c) * 1024;
  float x1 = src[(2 * i) * 32 + 2 * j]         * 0.5f; // even row, even col
  float x2 = src[(2 * i + 1) * 32 + 2 * j]     * 0.5f; // odd  row, even col
  float x3 = src[(2 * i) * 32 + 2 * j + 1]     * 0.5f; // even row, odd  col
  float x4 = src[(2 * i + 1) * 32 + 2 * j + 1] * 0.5f; // odd  row, odd  col
  float* dst = (f ? f2f : f1f) + (size_t)b * 512 * 256 + (size_t)c * 256 + i * 16 + j;
  dst[0]        =  x1 + x2 + x3 + x4;
  dst[128*256]  = -x1 - x2 + x3 + x4;
  dst[256*256]  = -x1 + x2 - x3 + x4;
  dst[384*256]  =  x1 - x2 - x3 + x4;
}

// Kernel 10: frequency gates m1f,m2f per (b, pixel<256) over 1024 channels
__global__ void k_m1m2f(const float* __restrict__ f1f, const float* __restrict__ f2f,
                        const float* __restrict__ g1, const float* __restrict__ g2,
                        float* __restrict__ m1f, float* __restrict__ m2f){
  int idx = blockIdx.x * blockDim.x + threadIdx.x; // 32*256
  int b = idx >> 8, p = idx & 255;
  const float* p1 = f1f + (size_t)b * 512 * 256 + p;
  const float* p2 = f2f + (size_t)b * 512 * 256 + p;
  float d1 = 0.f, d2 = 0.f;
  for (int c = 0; c < 512; ++c){
    float x1 = p1[(size_t)c * 256], x2 = p2[(size_t)c * 256];
    d1 += x1 * g1[c] + x2 * g1[512 + c];
    d2 += x1 * g2[c] + x2 * g2[512 + c];
  }
  m1f[idx] = sigmoidf(d1);
  m2f[idx] = sigmoidf(d2);
}

// Kernel 11: pooledf[b][ch<1024] = mean over 256 pixels
__global__ void k_poolf(const float* __restrict__ f1f, const float* __restrict__ f2f,
                        float* __restrict__ pooledf){
  int idx = blockIdx.x * blockDim.x + threadIdx.x; // 32*1024
  int b = idx >> 10, ch = idx & 1023;
  const float* src = (ch < 512 ? f1f + ((size_t)b * 512 + ch) * 256
                               : f2f + ((size_t)b * 512 + (ch - 512)) * 256);
  float s = 0.f;
  for (int p = 0; p < 256; ++p) s += src[p];
  pooledf[idx] = s * (1.0f / 256.0f);
}

// =====================================================================
// Kernel 12: build bf16 conv input cat[b][384][32][32]:
//   [0..127]   = SE * m * att    (i_spa)
//   [128..255] = iwt(SEf * mf * fre)  (i_fre)
//   [256..383] = downsampled frame
// =====================================================================
__global__ void k_cat(const float* __restrict__ att, const float* __restrict__ fre,
                      const float* __restrict__ fs,
                      const float* __restrict__ outg,  const float* __restrict__ m,
                      const float* __restrict__ outgf, const float* __restrict__ mf,
                      unsigned short* __restrict__ cat){
  int idx = blockIdx.x * blockDim.x + threadIdx.x; // 32*128*256 = 2^20
  int j = idx & 15, i = (idx >> 4) & 15;
  int c = (idx >> 8) & 127;
  int b = idx >> 15;
  size_t base = (size_t)b * 384 * 1024;
  const float* ap = att + ((size_t)b * 128 + c) * 1024;
  const float* fp = fs  + ((size_t)b * 128 + c) * 1024;
  float gq = outg[(size_t)b * 128 + c];
#pragma unroll
  for (int r = 0; r < 2; ++r)
#pragma unroll
    for (int t = 0; t < 2; ++t){
      int pix = (2 * i + r) * 32 + (2 * j + t);
      cat[base + (size_t)c * 1024 + pix]         = f2bf(gq * m[(size_t)b * 1024 + pix] * ap[pix]);
      cat[base + (size_t)(256 + c) * 1024 + pix] = f2bf(fp[pix]);
    }
  // IWT for channel c at block (i,j)
  int pf = i * 16 + j;
  float mm = mf[(size_t)b * 256 + pf];
  const float* fr = fre + (size_t)b * 512 * 256 + (size_t)c * 256 + pf;
  const float* gf = outgf + (size_t)b * 512;
  float v1 = gf[c]       * mm * fr[0]       * 0.5f;
  float v2 = gf[128 + c] * mm * fr[128*256] * 0.5f;
  float v3 = gf[256 + c] * mm * fr[256*256] * 0.5f;
  float v4 = gf[384 + c] * mm * fr[384*256] * 0.5f;
  float va = v1 - v2 - v3 + v4;
  float vb = v1 - v2 + v3 - v4;
  float vc = v1 + v2 - v3 - v4;
  float vd = v1 + v2 + v3 + v4;
  unsigned short* fc = cat + base + (size_t)(128 + c) * 1024;
  fc[(2 * i) * 32 + 2 * j]         = f2bf(va);
  fc[(2 * i) * 32 + 2 * j + 1]     = f2bf(vc);
  fc[(2 * i + 1) * 32 + 2 * j]     = f2bf(vb);
  fc[(2 * i + 1) * 32 + 2 * j + 1] = f2bf(vd);
}

// =====================================================================
// Kernel 13: 3x3 conv, 384->128 ch, as implicit WMMA GEMM (K = 9*384)
// grid = B*8 strips of 128 pixels; block = 8 waves (each: 16 out-chan x 8 pixel tiles)
// =====================================================================
DEVFN v16bf load_B_conv(const unsigned short* in, int p0, int cc, int dh, int dw, int lane){
  int hl  = lane >> 4;
  int pix = p0 + (lane & 15);
  int r  = (pix >> 5) + dh;
  int cw = (pix & 31) + dw;
  bool ok = (r >= 0 && r < 32 && cw >= 0 && cw < 32);
  int off = r * 32 + cw;
  BfPack t;
  const unsigned short* p = in + (size_t)(cc + hl * 16) * 1024 + off;
#pragma unroll
  for (int i = 0; i < 16; ++i) t.s[i] = ok ? p[(size_t)i * 1024] : (unsigned short)0;
  return t.v;
}

__global__ void k_conv(const unsigned short* __restrict__ cat,
                       const unsigned short* __restrict__ wb,
                       float* __restrict__ out){
  int lane  = threadIdx.x & 31;
  int wv    = threadIdx.x >> 5;
  int strip = blockIdx.x & 7;
  int b     = blockIdx.x >> 3;
  const unsigned short* in = cat + (size_t)b * 384 * 1024;
  int o0 = wv * 16;
  for (int pt = 0; pt < 8; ++pt){
    int p0 = strip * 128 + pt * 16;
    v8f acc = {0.f,0.f,0.f,0.f,0.f,0.f,0.f,0.f};
    for (int tap = 0; tap < 9; ++tap){
      int dh = tap / 3 - 1, dw = tap % 3 - 1;
      const unsigned short* wt = wb + (size_t)tap * 128 * 384;
      __builtin_prefetch((const void*)(wt + (size_t)o0 * 384), 0, 2);
      for (int cc = 0; cc < 384; cc += 32){
        v16bf a  = load_A_rowmajor(wt, 384, o0, cc, lane);
        v16bf bt = load_B_conv(in, p0, cc, dh, dw, lane);
        acc = wmma_bf16(a, bt, acc);
      }
    }
    float* D  = out + (size_t)b * 128 * 1024;
    int col   = p0 + (lane & 15);
    int rbase = o0 + ((lane >> 4) << 3);
#pragma unroll
    for (int r = 0; r < 8; ++r) D[(size_t)(rbase + r) * 1024 + col] = acc[r];
  }
}

// Kernel 14: per-channel batchnorm statistics. grid = 2*128 blocks, 256 threads.
__global__ void k_bnstats(const float* __restrict__ conv, float* __restrict__ mean,
                          float* __restrict__ rstd){
  int c = blockIdx.x & 127;
  int s = blockIdx.x >> 7;
  const float* src = conv + (size_t)s * 32 * 128 * 1024;
  float sum = 0.f, sq = 0.f;
  for (int i = threadIdx.x; i < 32 * 1024; i += 256){
    int b = i >> 10, p = i & 1023;
    float v = src[((size_t)b * 128 + c) * 1024 + p];
    sum += v; sq += v * v;
  }
  __shared__ float sh[512];
  sh[threadIdx.x] = sum; sh[256 + threadIdx.x] = sq;
  __syncthreads();
  for (int st = 128; st > 0; st >>= 1){
    if (threadIdx.x < st){
      sh[threadIdx.x]       += sh[threadIdx.x + st];
      sh[256 + threadIdx.x] += sh[256 + threadIdx.x + st];
    }
    __syncthreads();
  }
  if (threadIdx.x == 0){
    float mval = sh[0] * (1.0f / 32768.0f);
    float vv   = sh[256] * (1.0f / 32768.0f) - mval * mval;
    mean[blockIdx.x] = mval;
    rstd[blockIdx.x] = rsqrtf(vv + 1e-5f);
  }
}

// Kernel 15: BN + leaky-ReLU + bilinear 2x upsample (0.75/0.25, edge clamp) -> d_out
__global__ void k_bn_up(const float* __restrict__ conv,
                        const float* __restrict__ mean, const float* __restrict__ rstd,
                        const float* __restrict__ g1, const float* __restrict__ b1,
                        const float* __restrict__ g2, const float* __restrict__ b2,
                        float* __restrict__ out){
  unsigned int idx = blockIdx.x * blockDim.x + threadIdx.x; // 2*32*128*4096 = 2^25
  int ox = idx & 63, oy = (idx >> 6) & 63;
  int c  = (idx >> 12) & 127;
  int b  = (idx >> 19) & 31;
  int s  = idx >> 24;
  const float* src = conv + (size_t)s * 32 * 128 * 1024 + ((size_t)b * 128 + c) * 1024;
  float mval = mean[s * 128 + c], rv = rstd[s * 128 + c];
  float gg = s ? g2[c] : g1[c];
  float bb = s ? b2[c] : b1[c];
  float sy = oy * 0.5f - 0.25f;
  float sx = ox * 0.5f - 0.25f;
  int y0 = (int)floorf(sy); float wy = sy - y0;
  int x0 = (int)floorf(sx); float wx = sx - x0;
  int y0c = max(y0, 0), y1c = min(y0 + 1, 31);
  int x0c = max(x0, 0), x1c = min(x0 + 1, 31);
  auto val = [&](int yy, int xx){
    float v = src[yy * 32 + xx];
    return lreluf((v - mval) * rv * gg + bb);
  };
  float v00 = val(y0c, x0c), v01 = val(y0c, x1c);
  float v10 = val(y1c, x0c), v11 = val(y1c, x1c);
  out[idx] = (1.f - wy) * ((1.f - wx) * v00 + wx * v01) +
             wy * ((1.f - wx) * v10 + wx * v11);
}

// =====================================================================
extern "C" void kernel_launch(void* const* d_in, const int* in_sizes, int n_in,
                              void* d_out, int out_size, void* d_ws, size_t ws_size,
                              hipStream_t stream){
  const float* frame1  = (const float*)d_in[0];
  const float* frame2  = (const float*)d_in[1];
  const float* W_lin   = (const float*)d_in[2];
  const float* gate1_w = (const float*)d_in[3];
  const float* gate2_w = (const float*)d_in[4];
  const float* fc1_w   = (const float*)d_in[5];
  const float* fc1_b   = (const float*)d_in[6];
  const float* fc2_w   = (const float*)d_in[7];
  const float* fc2_b   = (const float*)d_in[8];
  const float* fc3_w   = (const float*)d_in[9];
  const float* fc3_b   = (const float*)d_in[10];
  const float* fc4_w   = (const float*)d_in[11];
  const float* fc4_b   = (const float*)d_in[12];
  const float* gate1f_w= (const float*)d_in[13];
  const float* gate2f_w= (const float*)d_in[14];
  const float* fc1f_w  = (const float*)d_in[15];
  const float* fc1f_b  = (const float*)d_in[16];
  const float* fc2f_w  = (const float*)d_in[17];
  const float* fc2f_b  = (const float*)d_in[18];
  const float* fc3f_w  = (const float*)d_in[19];
  const float* fc3f_b  = (const float*)d_in[20];
  const float* fc4f_w  = (const float*)d_in[21];
  const float* fc4f_b  = (const float*)d_in[22];
  const float* conv1_w = (const float*)d_in[23];
  const float* conv2_w = (const float*)d_in[24];
  const float* bn1_g   = (const float*)d_in[25];
  const float* bn1_b   = (const float*)d_in[26];
  const float* bn2_g   = (const float*)d_in[27];
  const float* bn2_b   = (const float*)d_in[28];

  char* ws = (char*)d_ws;
  constexpr size_t MB = 1024ull * 1024ull;
  constexpr size_t KB = 1024ull;

  float*          f1s   = (float*)(ws + 0);            // 16 MB
  float*          f2s   = (float*)(ws + 16  * MB);     // 16 MB
  unsigned short* f1b   = (unsigned short*)(ws + 32 * MB);  // 8 MB
  unsigned short* f2b   = (unsigned short*)(ws + 40 * MB);  // 8 MB
  unsigned short* g     = (unsigned short*)(ws + 48 * MB);  // 8 MB
  float*          Amat  = (float*)(ws + 56  * MB);     // 128 MB
  float*          cmax  = (float*)(ws + 184 * MB);            // 128 KB
  float*          csum  = (float*)(ws + 184 * MB + 128 * KB);
  float*          rmax  = (float*)(ws + 184 * MB + 256 * KB);
  float*          rsum  = (float*)(ws + 184 * MB + 384 * KB);
  float*          f1att = (float*)(ws + 185 * MB);     // 16 MB
  float*          f2att = (float*)(ws + 201 * MB);     // 16 MB
  float*          m1    = (float*)(ws + 217 * MB);            // 128 KB
  float*          m2    = (float*)(ws + 217 * MB + 128 * KB); // 128 KB
  float*          pooled= (float*)(ws + 217 * MB + 256 * KB); // 32 KB
  float*          out_e = (float*)(ws + 217 * MB + 288 * KB); // 16 KB
  float*          out_q = (float*)(ws + 217 * MB + 304 * KB); // 16 KB
  float*          f1fre = (float*)(ws + 218 * MB);     // 16 MB
  float*          f2fre = (float*)(ws + 234 * MB);     // 16 MB
  float*          m1f   = (float*)(ws + 250 * MB);            // 32 KB
  float*          m2f   = (float*)(ws + 250 * MB + 32 * KB);  // 32 KB
  float*          pooledf=(float*)(ws + 250 * MB + 64 * KB);  // 128 KB
  float*          out_ef= (float*)(ws + 250 * MB + 192 * KB); // 64 KB
  float*          out_qf= (float*)(ws + 250 * MB + 256 * KB); // 64 KB
  unsigned short* cat1  = (unsigned short*)(ws + 251 * MB); // 24 MB
  unsigned short* cat2  = (unsigned short*)(ws + 275 * MB); // 24 MB
  unsigned short* Wlb   = (unsigned short*)(ws + 299 * MB);            // 32 KB
  unsigned short* w1b   = (unsigned short*)(ws + 299 * MB + 32 * KB);  // 864 KB
  unsigned short* w2b   = (unsigned short*)(ws + 299 * MB + 928 * KB); // 864 KB
  float*          conv1 = (float*)(ws + 301 * MB);     // 16 MB  (conv2 contiguous after)
  float*          bnm   = (float*)(ws + 333 * MB);            // 1 KB
  float*          bnr   = (float*)(ws + 333 * MB + 4 * KB);   // 1 KB

  // 0) weight packing
  {
    int total = 128 * 128 + 2 * 9 * 128 * 384;
    k_convw<<<(total + 255) / 256, 256, 0, stream>>>(W_lin, conv1_w, conv2_w, Wlb, w1b, w2b);
  }
  // 1) downsample + bf16 copies
  k_down<<<(2 * 32 * 128 * 1024) / 256, 256, 0, stream>>>(frame1, frame2, f1s, f2s, f1b, f2b);
  // 2) g = W_lin * f1
  k_lin<<<(32 * 8 * 64) / 4, 128, 0, stream>>>(Wlb, f1b, g);
  // 3) affinity matrix A
  k_attA<<<(32 * 64 * 64) / 4, 128, 0, stream>>>(g, f2b, Amat);
  // 4) softmax stats
  k_red_col<<<(32 * 1024) / 256, 256, 0, stream>>>(Amat, cmax, csum);
  k_red_row<<<(32 * 1024) / 256, 256, 0, stream>>>(Amat, rmax, rsum);
  // 5) attention apply (exp fused into WMMA B loads)
  k_att_apply<0><<<(32 * 8 * 64) / 4, 128, 0, stream>>>(f1b, Amat, cmax, csum, f2att);
  k_att_apply<1><<<(32 * 8 * 64) / 4, 128, 0, stream>>>(f2b, Amat, rmax, rsum, f1att);
  // 6) spatial gates + SE
  k_m1m2<<<(32 * 1024) / 256, 256, 0, stream>>>(f1att, f2att, gate1_w, gate2_w, m1, m2);
  k_pool<<<(32 * 256) / 256, 256, 0, stream>>>(f1att, f2att, pooled);
  k_se<<<32, 256, 0, stream>>>(pooled, 256, fc1_w, fc1_b, fc2_w, fc2_b, 128, out_e);
  k_se<<<32, 256, 0, stream>>>(pooled, 256, fc3_w, fc3_b, fc4_w, fc4_b, 128, out_q);
  // 7) DWT + frequency gates + SE
  k_dwt<<<(2 * 32 * 128 * 256) / 256, 256, 0, stream>>>(f1att, f2att, f1fre, f2fre);
  k_m1m2f<<<(32 * 256) / 256, 256, 0, stream>>>(f1fre, f2fre, gate1f_w, gate2f_w, m1f, m2f);
  k_poolf<<<(32 * 1024) / 256, 256, 0, stream>>>(f1fre, f2fre, pooledf);
  k_se<<<32, 256, 0, stream>>>(pooledf, 1024, fc1f_w, fc1f_b, fc2f_w, fc2f_b, 512, out_ef);
  k_se<<<32, 256, 0, stream>>>(pooledf, 1024, fc3f_w, fc3f_b, fc4f_w, fc4f_b, 512, out_qf);
  // 8) build conv inputs (i_spa | iwt(i_fre) | f), bf16
  k_cat<<<(32 * 128 * 256) / 256, 256, 0, stream>>>(f1att, f1fre, f1s, out_q, m1, out_qf, m1f, cat1);
  k_cat<<<(32 * 128 * 256) / 256, 256, 0, stream>>>(f2att, f2fre, f2s, out_e, m2, out_ef, m2f, cat2);
  // 9) WMMA implicit-GEMM convolutions
  k_conv<<<32 * 8, 256, 0, stream>>>(cat1, w1b, conv1);
  k_conv<<<32 * 8, 256, 0, stream>>>(cat2, w2b, conv1 + (size_t)32 * 128 * 1024);
  // 10) batchnorm statistics
  k_bnstats<<<2 * 128, 256, 0, stream>>>(conv1, bnm, bnr);
  // 11) BN + lrelu + bilinear upsample to d_out
  k_bn_up<<<(2 * 32 * 128 * 4096) / 256, 256, 0, stream>>>(conv1, bnm, bnr,
      bn1_g, bn1_b, bn2_g, bn2_b, (float*)d_out);
}
